// WireframeLoss_16621523436033
// MI455X (gfx1250) — compile-verified, matrix operand order checked by reading, not run on hardware
//
#include <hip/hip_runtime.h>
#include <hip/hip_bf16.h>
#include <math.h>

// Problem constants (match the reference).
#define BATCH 64
#define NVERT 128
#define NEDGE 8128

typedef __attribute__((ext_vector_type(16))) _Float16 v16h;
typedef __attribute__((ext_vector_type(8)))  float    v8f;

// ---------------------------------------------------------------------------
// Wave32 sum-reduction via the matrix unit: D = A(16x32 f16) x ones(32x16) + 0
// with f32 accumulation.  Each lane's partial x is split into two f16 terms
// (h1 = fp16(x), h2 = fp16(x - h1)) placed at K=0,1 so the f32 accumulator
// recovers x to ~2^-22 relative error.  A-layout: lane m holds K=0,1 of row m,
// lane m+16 holds K=8,9 of row m  =>  rowsum[m] = x[m] + x[m+16].
// C/D layout: lane 0 c[0..7] = rowsum[0..7], lane 16 c[0..7] = rowsum[8..15].
// Lanes 0 and 16 each issue one global_atomic_add_f32.
// Call convergently (EXEC must be all ones for WMMA).
// ---------------------------------------------------------------------------
__device__ __forceinline__ void wave32_wmma_reduce_atomic(float x, float* acc) {
  v16h a = {};
  _Float16 h1 = (_Float16)x;
  _Float16 h2 = (_Float16)(x - (float)h1);
  a[0] = h1;
  a[1] = h2;
  v16h b;
#pragma unroll
  for (int i = 0; i < 16; ++i) b[i] = (_Float16)1.0f;
  v8f c = {};
  c = __builtin_amdgcn_wmma_f32_16x16x32_f16(false, a, false, b,
                                             (short)0, c, false, false);
  float s = c[0] + c[1] + c[2] + c[3] + c[4] + c[5] + c[6] + c[7];
  unsigned lane = threadIdx.x & 31u;
  if (lane == 0u || lane == 16u) atomicAdd(acc, s);
}

// ---------------------------------------------------------------------------
// ws[0] = smooth-L1 sum, ws[1] = existence BCE sum, ws[2] = edge BCE sum.
// ---------------------------------------------------------------------------
__global__ void k_zero(float* ws) {
  if (threadIdx.x < 8) ws[threadIdx.x] = 0.0f;
}

// BCE with float targets (edge loss). torch-style clamp of log at -100.
__global__ void k_bce_f(const float* __restrict__ p, const float* __restrict__ t,
                        int n, float* acc) {
  int stride = gridDim.x * blockDim.x;
  float local = 0.0f;
  for (int i = blockIdx.x * blockDim.x + threadIdx.x; i < n; i += stride) {
    __builtin_prefetch(&p[i + stride], 0, 0);   // global_prefetch_b8
    __builtin_prefetch(&t[i + stride], 0, 0);
    float pp = p[i], tt = t[i];
    float lp = fmaxf(logf(pp), -100.0f);
    float lq = fmaxf(logf(1.0f - pp), -100.0f);
    local += -(tt * lp + (1.0f - tt) * lq);
  }
  wave32_wmma_reduce_atomic(local, acc);
}

// BCE with int targets (existence loss).
__global__ void k_bce_i(const float* __restrict__ p, const int* __restrict__ t,
                        int n, float* acc) {
  int stride = gridDim.x * blockDim.x;
  float local = 0.0f;
  for (int i = blockIdx.x * blockDim.x + threadIdx.x; i < n; i += stride) {
    float pp = p[i], tt = (float)t[i];
    float lp = fmaxf(logf(pp), -100.0f);
    float lq = fmaxf(logf(1.0f - pp), -100.0f);
    local += -(tt * lp + (1.0f - tt) * lq);
  }
  wave32_wmma_reduce_atomic(local, acc);
}

// ---------------------------------------------------------------------------
// Hungarian matching (Jonker-Volgenant with row potentials, mirroring _lap)
// + SmoothL1 over matched pairs.  One wave32 workgroup per batch sample:
// single-wave workgroups make s_barrier a NOP (ISA 3.1), so the serial
// augmenting-path loop costs only shuffles + LDS.  Lane L owns columns
// j = 1 + L + 32q, q = 0..3; v/minv/used for owned columns stay in registers.
// Cost entries are recomputed on the fly from LDS-cached vertices.
// ---------------------------------------------------------------------------
#define LAP_INF 1e30f

__global__ void __launch_bounds__(32) k_match(
    const float* __restrict__ pv, const float* __restrict__ pe,
    const float* __restrict__ tv, const int* __restrict__ counts,
    float* acc_sl1) {
  const int b = blockIdx.x;
  const int lane = threadIdx.x;  // 0..31

  __shared__ float sU[NVERT + 1];
  __shared__ int   sP[NVERT + 1];
  __shared__ int   sWay[NVERT + 1];
  __shared__ float sPvx[NVERT], sPvy[NVERT], sPvz[NVERT];
  __shared__ float sTvx[NVERT], sTvy[NVERT], sTvz[NVERT];
  __shared__ float sPe[NVERT], sEp[NVERT];

  const int c = counts[b];

#pragma unroll
  for (int q = 0; q < 4; ++q) {
    int i = lane + 32 * q;
    sPvx[i] = pv[(b * NVERT + i) * 3 + 0];
    sPvy[i] = pv[(b * NVERT + i) * 3 + 1];
    sPvz[i] = pv[(b * NVERT + i) * 3 + 2];
    sTvx[i] = tv[(b * NVERT + i) * 3 + 0];
    sTvy[i] = tv[(b * NVERT + i) * 3 + 1];
    sTvz[i] = tv[(b * NVERT + i) * 3 + 2];
    float e = pe[b * NVERT + i];
    sPe[i] = e;                    // cost to a dummy column
    sEp[i] = fabsf(e - 1.0f);      // existence-match penalty to a real column
    sU[i] = 0.0f; sP[i] = 0; sWay[i] = 0;
  }
  if (lane == 0) { sU[NVERT] = 0.0f; sP[NVERT] = 0; sWay[NVERT] = 0; }
  __syncthreads();

  float vcol[4];                   // column potentials v[j] (owned columns)
  float minv[4];
#pragma unroll
  for (int q = 0; q < 4; ++q) vcol[q] = 0.0f;

  for (int i = 1; i <= NVERT; ++i) {
    if (lane == 0) sP[0] = i;
#pragma unroll
    for (int q = 0; q < 4; ++q) minv[q] = LAP_INF;
    int usedm = 0;
    int j0 = 0;
    __syncthreads();

    while (true) {
      // used[j0] = true (column 0 handled implicitly: always used).
      if (j0 >= 1) {
        int cc0 = j0 - 1;
        if ((cc0 & 31) == lane) usedm |= 1 << (cc0 >> 5);
      }
      int   i0  = sP[j0];
      float ui0 = sU[i0];
      int   r   = i0 - 1;
      float px = sPvx[r], py = sPvy[r], pz = sPvz[r];
      float ep = sEp[r],  dm = sPe[r];

      float bestv = LAP_INF;
      int   bestj = 0x7fffffff;
#pragma unroll
      for (int q = 0; q < 4; ++q) {
        if (!(usedm & (1 << q))) {
          int j  = 1 + lane + 32 * q;
          int cc = j - 1;
          float cost = (cc < c)
              ? fabsf(px - sTvx[cc]) + fabsf(py - sTvy[cc]) +
                fabsf(pz - sTvz[cc]) + ep
              : dm;
          float cur = cost - ui0 - vcol[q];
          if (cur < minv[q]) { minv[q] = cur; sWay[j] = j0; }
          if (minv[q] < bestv || (minv[q] == bestv && j < bestj)) {
            bestv = minv[q]; bestj = j;
          }
        }
      }
      // Wave argmin (ties -> lowest column index, matching np.argmin).
#pragma unroll
      for (int off = 16; off >= 1; off >>= 1) {
        float ov = __shfl_xor(bestv, off, 32);
        int   oj = __shfl_xor(bestj, off, 32);
        if (ov < bestv || (ov == bestv && oj < bestj)) { bestv = ov; bestj = oj; }
      }
      float delta = bestv;
      int   j1    = bestj;

      // u[p[used]] += delta; v[used] -= delta; minv[free] -= delta.
      if (lane == 0) sU[sP[0]] += delta;   // column 0 is always in the used set
#pragma unroll
      for (int q = 0; q < 4; ++q) {
        int j = 1 + lane + 32 * q;
        if (usedm & (1 << q)) {
          sU[sP[j]] += delta;              // p[] values are distinct -> no race
          vcol[q] -= delta;
        } else {
          minv[q] -= delta;
        }
      }
      __syncthreads();                     // order sU writes vs. next-iter reads
      j0 = j1;
      if (sP[j0] == 0) break;              // reached an unmatched column
    }

    // Augment along the alternating path (serial, lane 0).
    if (lane == 0) {
      int j0a = j0;
      while (j0a != 0) {
        int j1a = sWay[j0a];
        sP[j0a] = sP[j1a];
        j0a = j1a;
      }
    }
    __syncthreads();
  }

  // SmoothL1 (beta=1) over matches to real targets: column cc -> row sP[cc+1]-1.
  float local = 0.0f;
  for (int cc = lane; cc < c; cc += 32) {
    int r = sP[cc + 1] - 1;
    float dx = fabsf(sPvx[r] - sTvx[cc]);
    float dy = fabsf(sPvy[r] - sTvy[cc]);
    float dz = fabsf(sPvz[r] - sTvz[cc]);
    local += (dx < 1.0f) ? 0.5f * dx * dx : dx - 0.5f;
    local += (dy < 1.0f) ? 0.5f * dy * dy : dy - 0.5f;
    local += (dz < 1.0f) ? 0.5f * dz * dz : dz - 0.5f;
  }
  wave32_wmma_reduce_atomic(local, acc_sl1);
}

// ---------------------------------------------------------------------------
// Finalize: normalize the three sums and emit (total, vertex, exist, edge).
// ---------------------------------------------------------------------------
__global__ void k_final(const int* __restrict__ counts,
                        const float* __restrict__ ws, float* out) {
  if (threadIdx.x == 0) {
    int m = 0;
    for (int b = 0; b < BATCH; ++b) m += counts[b];
    float vloss = ws[0] / (3.0f * (float)m);
    float eloss = ws[1] / (float)(BATCH * NVERT);
    float gloss = ws[2] / (float)(BATCH * NEDGE);
    out[0] = vloss + eloss + gloss;  // weights are all 1.0
    out[1] = vloss;
    out[2] = eloss;
    out[3] = gloss;
  }
}

extern "C" void kernel_launch(void* const* d_in, const int* in_sizes, int n_in,
                              void* d_out, int out_size, void* d_ws, size_t ws_size,
                              hipStream_t stream) {
  (void)in_sizes; (void)n_in; (void)out_size; (void)ws_size;
  const float* pred_vertices  = (const float*)d_in[0];
  const float* pred_existence = (const float*)d_in[1];
  const float* pred_edges     = (const float*)d_in[2];
  const float* targ_vertices  = (const float*)d_in[3];
  const int*   targ_existence = (const int*)d_in[4];
  const float* edge_labels    = (const float*)d_in[5];
  const int*   vertex_counts  = (const int*)d_in[6];
  float* out = (float*)d_out;
  float* ws  = (float*)d_ws;   // ws[0]=sl1 sum, ws[1]=exist sum, ws[2]=edge sum

  hipLaunchKernelGGL(k_zero, dim3(1), dim3(32), 0, stream, ws);
  hipLaunchKernelGGL(k_bce_f, dim3(256), dim3(256), 0, stream,
                     pred_edges, edge_labels, BATCH * NEDGE, ws + 2);
  hipLaunchKernelGGL(k_bce_i, dim3(16), dim3(256), 0, stream,
                     pred_existence, targ_existence, BATCH * NVERT, ws + 1);
  hipLaunchKernelGGL(k_match, dim3(BATCH), dim3(32), 0, stream,
                     pred_vertices, pred_existence, targ_vertices,
                     vertex_counts, ws + 0);
  hipLaunchKernelGGL(k_final, dim3(1), dim3(32), 0, stream,
                     vertex_counts, ws, out);
}
